// FermiNet_34806414967410
// MI455X (gfx1250) — matrix operand order repaired and checked by reading, not compiled, not run on hardware
//
#include <hip/hip_runtime.h>
#include <math.h>

// ---------------------------------------------------------------------------
// FermiNet-style network on MI455X (gfx1250), wave32 + WMMA.
// Sizes: N=1024, DIM=3, H1=64, H2=16, NF=5, K=4, L=10. F0=105, F=304.
// ---------------------------------------------------------------------------

typedef __attribute__((ext_vector_type(16))) _Float16 v16h;
typedef __attribute__((ext_vector_type(8)))  _Float16 v8h;
typedef __attribute__((ext_vector_type(8)))  float    v8f;

#define DEVINL __device__ __forceinline__

__constant__ const float kL   = 10.0f;
__constant__ const float kPI2 = 6.283185307179586f;

DEVINL v8f wmma16(v16h a, v16h b, v8f c) {
  // D = A(16x32 f16) * B(32x16 f16) + C(16x16 f32)
  return __builtin_amdgcn_wmma_f32_16x16x32_f16(
      /*neg_a=*/false, a, /*neg_b=*/false, b,
      /*c_mod=*/(short)0, c, /*reuse_a=*/false, /*reuse_b=*/false);
}

DEVINL v16h pack_a(v8h lo, v8h hi) {
  v16h r;
#pragma unroll
  for (int e = 0; e < 8; ++e) { r[e] = lo[e]; r[8 + e] = hi[e]; }
  return r;
}

// branchless tanh: 1 - 2/(exp(2x)+1) using raw v_exp_f32 + v_rcp_f32
DEVINL float fast_tanh(float x) {
  const float t = __expf(2.0f * x);
  return 1.0f - 2.0f * __builtin_amdgcn_rcpf(t + 1.0f);
}

// ---------------------------------------------------------------------------
// zero scratch
// ---------------------------------------------------------------------------
__global__ void k_zero(float* __restrict__ p, int n) {
  int i = blockIdx.x * blockDim.x + threadIdx.x;
  if (i < n) p[i] = 0.0f;
}

// ---------------------------------------------------------------------------
// Pad + transpose weights to f16:  W[F][Nn] -> WT[Nn][Kp], zero-padded K.
// Makes every WMMA B-fragment two contiguous 16B loads.
// ---------------------------------------------------------------------------
__global__ void k_wpad(const float* __restrict__ W, _Float16* __restrict__ WT,
                       int F, int Nn, int Kp) {
  const int idx = blockIdx.x * blockDim.x + threadIdx.x;
  if (idx >= Nn * Kp) return;
  const int n = idx / Kp, k = idx - n * Kp;
  WT[idx] = (_Float16)((k < F) ? W[k * Nn + n] : 0.0f);
}

// ---------------------------------------------------------------------------
// Fused pair-feature + 3-layer h2 MLP. Never materializes h2: accumulates
// per-(split, j, channel) sums of raw features and of h2 after each layer.
// grid = (64 j-tiles, 32 i-chunks), block = 128 (4 waves, 8 i's per wave).
// wt2: padded f16 weights [3 layers][16 n][32 k].
// g2raw: [3][1024][32] sums (31 valid). g2h: [3 depths][3 splits][1024][16].
// ---------------------------------------------------------------------------
__global__ __launch_bounds__(128) void k_h2(
    const float* __restrict__ x, const _Float16* __restrict__ wt2,
    const float* __restrict__ b0, const float* __restrict__ bseq,
    float* __restrict__ g2raw, float* __restrict__ g2h) {
  const int lane = threadIdx.x & 31;
  const int wave = threadIdx.x >> 5;
  const int h    = lane >> 4;   // half-wave
  const int q    = lane & 15;   // j-local row for A, out-channel for B/C
  const int jt   = blockIdx.x;
  const int j    = jt * 16 + q;
  const int i0   = blockIdx.y * 32 + wave * 8;
  const int split = (i0 < 512) ? 0 : (i0 < 768 ? 1 : 2);

  // B fragments: two contiguous v8h loads each from padded-transposed weights
  v16h B0, B1, B2;
  {
    const _Float16* p0 = wt2 + (0 * 16 + q) * 32 + h * 8;
    const _Float16* p1 = wt2 + (1 * 16 + q) * 32 + h * 8;
    const _Float16* p2 = wt2 + (2 * 16 + q) * 32 + h * 8;
    B0 = pack_a(*(const v8h*)p0, *(const v8h*)(p0 + 16));
    B1 = pack_a(*(const v8h*)p1, *(const v8h*)(p1 + 16));
    B2 = pack_a(*(const v8h*)p2, *(const v8h*)(p2 + 16));
  }
  const float bia0 = b0[q], bia1 = bseq[q], bia2 = bseq[16 + q];

  const float xj0 = x[j * 3 + 0], xj1 = x[j * 3 + 1], xj2 = x[j * 3 + 2];

  __shared__ _Float16 ftile[4][16][32];  // per-wave feature tile (row j, col K)
  __shared__ _Float16 xch[4][16][16];    // per-wave C->A relayout tile

  float raw[16];
#pragma unroll
  for (int e = 0; e < 16; ++e) raw[e] = 0.0f;
  v8f a1 = {}, a2 = {}, a3 = {};

  for (int t = 0; t < 8; ++t) {
    const int i = i0 + t;
    // ---- features for pair (i, j): ch[0]=r, then cos/sin(m*t_d), m=1..5 ----
    float ch[32];
    {
      const float wph = kPI2 / kL;
      const float t0 = (x[i * 3 + 0] - xj0) * wph;
      const float t1 = (x[i * 3 + 1] - xj1) * wph;
      const float t2 = (x[i * 3 + 2] - xj2) * wph;
      float s0, c0, s1, c1, s2, c2;
      __sincosf(t0, &s0, &c0);
      __sincosf(t1, &s1, &c1);
      __sincosf(t2, &s2, &c2);
      ch[0] = (i == j) ? 0.0f : sqrtf(s0 * s0 + s1 * s1 + s2 * s2);
      float cm0 = c0, cm1 = c1, cm2 = c2, sm0 = s0, sm1 = s1, sm2 = s2;
#pragma unroll
      for (int m = 0; m < 5; ++m) {
        ch[1 + 6 * m + 0] = cm0; ch[1 + 6 * m + 1] = cm1; ch[1 + 6 * m + 2] = cm2;
        ch[4 + 6 * m + 0] = sm0; ch[4 + 6 * m + 1] = sm1; ch[4 + 6 * m + 2] = sm2;
        float n0 = cm0 * c0 - sm0 * s0; sm0 = sm0 * c0 + cm0 * s0; cm0 = n0;
        float n1 = cm1 * c1 - sm1 * s1; sm1 = sm1 * c1 + cm1 * s1; cm1 = n1;
        float n2 = cm2 * c2 - sm2 * s2; sm2 = sm2 * c2 + cm2 * s2; cm2 = n2;
      }
      ch[31] = 0.0f;
    }
    // ---- store channels to LDS (packed), read back A-fragment slices ----
    {
      v8h p0, p1, p2, p3;
#pragma unroll
      for (int e = 0; e < 8; ++e) {
        p0[e] = (_Float16)ch[e];      p1[e] = (_Float16)ch[8 + e];
        p2[e] = (_Float16)ch[16 + e]; p3[e] = (_Float16)ch[24 + e];
      }
      *(v8h*)&ftile[wave][q][0]  = p0;
      *(v8h*)&ftile[wave][q][8]  = p1;
      *(v8h*)&ftile[wave][q][16] = p2;
      *(v8h*)&ftile[wave][q][24] = p3;
    }
    v8h alo = *(const v8h*)&ftile[wave][q][h * 8];
    v8h ahi = *(const v8h*)&ftile[wave][q][16 + h * 8];
    v16h A0 = pack_a(alo, ahi);
#pragma unroll
    for (int e = 0; e < 8; ++e) { raw[e] += (float)alo[e]; raw[8 + e] += (float)ahi[e]; }

    const v8f czero = {};
    // ---- layer 0 ----
    v8f t0r = wmma16(A0, B0, czero);
    v8f hA;
#pragma unroll
    for (int v = 0; v < 8; ++v) { hA[v] = fast_tanh(t0r[v] + bia0); a1[v] += hA[v]; }
#pragma unroll
    for (int v = 0; v < 8; ++v) xch[wave][v + h * 8][q] = (_Float16)hA[v];
    v8h zlo = {};
    v16h A1 = pack_a(*(const v8h*)&xch[wave][q][h * 8], zlo);
    // ---- layer 1 (+residual) ----
    v8f t1r = wmma16(A1, B1, czero);
    v8f hB;
#pragma unroll
    for (int v = 0; v < 8; ++v) { hB[v] = fast_tanh(t1r[v] + bia1) + hA[v]; a2[v] += hB[v]; }
#pragma unroll
    for (int v = 0; v < 8; ++v) xch[wave][v + h * 8][q] = (_Float16)hB[v];
    v16h A2 = pack_a(*(const v8h*)&xch[wave][q][h * 8], zlo);
    // ---- layer 2 (+residual) ----
    v8f t2r = wmma16(A2, B2, czero);
#pragma unroll
    for (int v = 0; v < 8; ++v) a3[v] += fast_tanh(t2r[v] + bia2) + hB[v];
  }

  // ---- flush accumulators ----
#pragma unroll
  for (int e = 0; e < 16; ++e) {
    const int k = (e < 8) ? (h * 8 + e) : (16 + h * 8 + (e - 8));
    if (k < 31) atomicAdd(&g2raw[(split * 1024 + j) * 32 + k], raw[e]);
  }
#pragma unroll
  for (int v = 0; v < 8; ++v) {
    const int row = jt * 16 + v + h * 8;
    atomicAdd(&g2h[((0 * 3 + split) * 1024 + row) * 16 + q], a1[v]);
    atomicAdd(&g2h[((1 * 3 + split) * 1024 + row) * 16 + q], a2[v]);
    atomicAdd(&g2h[((2 * 3 + split) * 1024 + row) * 16 + q], a3[v]);
  }
}

// ---------------------------------------------------------------------------
// build f for depth 0: [zeros(12) | g2raw means (3*31)] padded to 128, f16.
// ---------------------------------------------------------------------------
__global__ void k_buildf0(const float* __restrict__ g2raw, _Float16* __restrict__ f) {
  const int idx = blockIdx.x * blockDim.x + threadIdx.x;  // 1024*128
  const int row = idx >> 7, col = idx & 127;
  float v = 0.0f;
  if (col >= 12 && col < 105) {
    const int cc = col - 12, s = cc / 31, c = cc % 31;
    const float inv = (s == 0) ? (1.0f / 512.0f) : (1.0f / 256.0f);
    v = g2raw[(s * 1024 + row) * 32 + c] * inv;
  }
  f[idx] = (_Float16)v;
}

// column means of h1 per split -> g1[3*64]
__global__ void k_colmeans(const float* __restrict__ h1, float* __restrict__ g1) {
  const int t = threadIdx.x;
  if (t >= 192) return;
  const int s = t >> 6, c = t & 63;
  const int r0 = (s == 0) ? 0 : (s == 1 ? 512 : 768);
  const int r1 = (s == 0) ? 512 : (s == 1 ? 768 : 1024);
  float acc = 0.0f;
  for (int r = r0; r < r1; ++r) acc += h1[r * 64 + c];
  g1[t] = acc / (float)(r1 - r0);
}

// build f for depth >=1: [h1(64) | g1(3*64) | g2 means (3*16)] pad->320, f16.
__global__ void k_buildf(const float* __restrict__ h1, const float* __restrict__ g1,
                         const float* __restrict__ g2d, _Float16* __restrict__ f) {
  const int idx = blockIdx.x * blockDim.x + threadIdx.x;  // 1024*320
  const int row = idx / 320, col = idx - row * 320;
  float v = 0.0f;
  if (col < 64) v = h1[row * 64 + col];
  else if (col < 256) v = g1[col - 64];
  else if (col < 304) {
    const int cc = col - 256, s = cc >> 4, c = cc & 15;
    const float inv = (s == 0) ? (1.0f / 512.0f) : (1.0f / 256.0f);
    v = g2d[(s * 1024 + row) * 16 + c] * inv;
  }
  f[idx] = (_Float16)v;
}

// ---------------------------------------------------------------------------
// h1 GEMM: (1024 x Kp f16) @ WT[64][Kp] f16 + bias, tanh, optional residual.
// grid = 8 blocks x 256 threads = 64 waves = 64 M-tiles; each wave 4 N-tiles.
// ---------------------------------------------------------------------------
__global__ __launch_bounds__(256) void k_h1gemm(
    const _Float16* __restrict__ f, int Kp, const _Float16* __restrict__ WT,
    const float* __restrict__ bias, float* __restrict__ h1,
    _Float16* __restrict__ h1h, int residual) {
  const int lane = threadIdx.x & 31, wave = threadIdx.x >> 5;
  const int h = lane >> 4, q = lane & 15;
  const int mt = blockIdx.x * 8 + wave;
  const int row = mt * 16 + q;

  v8f acc[4];
#pragma unroll
  for (int nt = 0; nt < 4; ++nt) { v8f z = {}; acc[nt] = z; }

  const int kiters = Kp >> 5;
  for (int kt = 0; kt < kiters; ++kt) {
    const _Float16* ap = f + (size_t)row * Kp + kt * 32;
    v16h A = pack_a(*(const v8h*)(ap + h * 8), *(const v8h*)(ap + 16 + h * 8));
#pragma unroll
    for (int nt = 0; nt < 4; ++nt) {
      const _Float16* bp = WT + (size_t)(nt * 16 + q) * Kp + kt * 32 + h * 8;
      v16h B = pack_a(*(const v8h*)bp, *(const v8h*)(bp + 16));
      acc[nt] = wmma16(A, B, acc[nt]);
    }
  }
#pragma unroll
  for (int nt = 0; nt < 4; ++nt) {
#pragma unroll
    for (int v = 0; v < 8; ++v) {
      const int rr = mt * 16 + v + h * 8;
      const int n = nt * 16 + q;
      float val = fast_tanh(acc[nt][v] + bias[n]);
      if (residual) val += h1[rr * 64 + n];
      h1[rr * 64 + n] = val;
      h1h[rr * 64 + n] = (_Float16)val;
    }
  }
}

// ---------------------------------------------------------------------------
// z = hd @ final_w + final_b + x[512:]
// ---------------------------------------------------------------------------
__global__ void k_z(const float* __restrict__ h1, const float* __restrict__ fw,
                    const float* __restrict__ fb, const float* __restrict__ x,
                    float* __restrict__ z) {
  const int jd = blockIdx.x * blockDim.x + threadIdx.x;
  if (jd >= 512) return;
  float a0 = fb[0], a1 = fb[1], a2 = fb[2];
  for (int a = 0; a < 64; ++a) {
    const float hv = h1[(512 + jd) * 64 + a];
    a0 += hv * fw[a * 3 + 0]; a1 += hv * fw[a * 3 + 1]; a2 += hv * fw[a * 3 + 2];
  }
  z[jd * 3 + 0] = a0 + x[(512 + jd) * 3 + 0];
  z[jd * 3 + 1] = a1 + x[(512 + jd) * 3 + 1];
  z[jd * 3 + 2] = a2 + x[(512 + jd) * 3 + 2];
}

// r[i,jd] = || wrap(x_i - z_jd) ||
__global__ void k_r(const float* __restrict__ x, const float* __restrict__ z,
                    float* __restrict__ r) {
  const int idx = blockIdx.x * blockDim.x + threadIdx.x;
  if (idx >= 512 * 512) return;
  const int i = idx >> 9, jd = idx & 511;
  float s = 0.0f;
#pragma unroll
  for (int dd = 0; dd < 3; ++dd) {
    float d = x[i * 3 + dd] - z[jd * 3 + dd];
    d -= kL * rintf(d * (1.0f / kL));
    s += d * d;
  }
  r[idx] = sqrtf(s);
}

// ---------------------------------------------------------------------------
// tmp[k,i,b] = sum_a hu[i,a] * w_det[k,a,b]  (512x64 @ 64x64 per k, f16 out)
// wdT = padded/transposed w_det: [k][b][a].
// grid = (4 mtile-groups, 4 k), block 256 (8 waves = 32 M-tiles).
// ---------------------------------------------------------------------------
__global__ __launch_bounds__(256) void k_tmp(const _Float16* __restrict__ h1h,
                                             const _Float16* __restrict__ wdT,
                                             _Float16* __restrict__ tmp) {
  const int lane = threadIdx.x & 31, wave = threadIdx.x >> 5;
  const int h = lane >> 4, q = lane & 15;
  const int k = blockIdx.y;
  const int mt = blockIdx.x * 8 + wave;
  const int row = mt * 16 + q;  // i (hu rows 0..511)

  v8f acc[4];
#pragma unroll
  for (int nt = 0; nt < 4; ++nt) { v8f z = {}; acc[nt] = z; }

#pragma unroll
  for (int kt = 0; kt < 2; ++kt) {
    const _Float16* ap = h1h + row * 64 + kt * 32;
    v16h A = pack_a(*(const v8h*)(ap + h * 8), *(const v8h*)(ap + 16 + h * 8));
#pragma unroll
    for (int nt = 0; nt < 4; ++nt) {
      const _Float16* bp = wdT + ((size_t)k * 64 + nt * 16 + q) * 64 + kt * 32 + h * 8;
      v16h B = pack_a(*(const v8h*)bp, *(const v8h*)(bp + 16));
      acc[nt] = wmma16(A, B, acc[nt]);
    }
  }
#pragma unroll
  for (int nt = 0; nt < 4; ++nt)
#pragma unroll
    for (int v = 0; v < 8; ++v) {
      const int i = mt * 16 + v + h * 8;
      tmp[(size_t)(k * 512 + i) * 64 + nt * 16 + q] = (_Float16)acc[nt][v];
    }
}

// rowbias[k,i] = hu[i,:] . b_det[k,:] + 1
__global__ void k_rowbias(const float* __restrict__ h1, const float* __restrict__ bdet,
                          float* __restrict__ rb) {
  const int idx = blockIdx.x * blockDim.x + threadIdx.x;
  if (idx >= 2048) return;
  const int k = idx >> 9, i = idx & 511;
  float acc = 1.0f;
  for (int a = 0; a < 64; ++a) acc += h1[i * 64 + a] * bdet[k * 64 + a];
  rb[idx] = acc;
}

// ---------------------------------------------------------------------------
// Amat[k,i,j] = exp(-softplus(alpha_k)*r[i,j]) * (tmp_k @ hd^T + rowbias)
// grid = (32 mtiles, 4 k), block 256: wave w covers N-tiles 4w..4w+3.
// ---------------------------------------------------------------------------
__global__ __launch_bounds__(256) void k_amat(
    const _Float16* __restrict__ tmp, const _Float16* __restrict__ h1h,
    const float* __restrict__ rb, const float* __restrict__ r,
    const float* __restrict__ alpha, float* __restrict__ Aout) {
  const int lane = threadIdx.x & 31, wave = threadIdx.x >> 5;
  const int h = lane >> 4, q = lane & 15;
  const int mt = blockIdx.x, k = blockIdx.y;
  const float sa = __logf(1.0f + __expf(alpha[k]));  // softplus
  const int row = mt * 16 + q;  // i

  v8f acc[4];
#pragma unroll
  for (int nt = 0; nt < 4; ++nt) { v8f z = {}; acc[nt] = z; }

#pragma unroll
  for (int kt = 0; kt < 2; ++kt) {
    const _Float16* ap = tmp + (size_t)(k * 512 + row) * 64 + kt * 32;
    v16h A = pack_a(*(const v8h*)(ap + h * 8), *(const v8h*)(ap + 16 + h * 8));
#pragma unroll
    for (int nt = 0; nt < 4; ++nt) {
      const int j = (wave * 4 + nt) * 16 + q;  // B column
      const _Float16* bp = h1h + (size_t)(512 + j) * 64 + kt * 32 + h * 8;
      v16h B = pack_a(*(const v8h*)bp, *(const v8h*)(bp + 16));
      acc[nt] = wmma16(A, B, acc[nt]);
    }
  }
#pragma unroll
  for (int nt = 0; nt < 4; ++nt)
#pragma unroll
    for (int v = 0; v < 8; ++v) {
      const int i = mt * 16 + v + h * 8;
      const int j = (wave * 4 + nt) * 16 + q;
      const float phi = acc[nt][v] + rb[k * 512 + i];
      Aout[(size_t)(k * 512 + i) * 512 + j] = __expf(-sa * r[i * 512 + j]) * phi;
    }
}

// ---------------------------------------------------------------------------
// slogdet via LU w/ partial pivoting; one workgroup per k-matrix (512x512).
// ---------------------------------------------------------------------------
__global__ __launch_bounds__(256) void k_lu(float* __restrict__ Am,
                                            float* __restrict__ sld) {
  const int k = blockIdx.x, tid = threadIdx.x;
  float* A = Am + (size_t)k * 512 * 512;
  __shared__ float sv[256];
  __shared__ int si[256];
  __shared__ int piv;
  int swsign = 1;  // only tid 0's copy matters

  for (int c = 0; c < 512; ++c) {
    float best = -1.0f; int bi = c;
    for (int rr = c + tid; rr < 512; rr += 256) {
      const float v = fabsf(A[rr * 512 + c]);
      if (v > best) { best = v; bi = rr; }
    }
    sv[tid] = best; si[tid] = bi;
    __syncthreads();
    for (int off = 128; off > 0; off >>= 1) {
      if (tid < off && sv[tid + off] > sv[tid]) { sv[tid] = sv[tid + off]; si[tid] = si[tid + off]; }
      __syncthreads();
    }
    if (tid == 0) { piv = si[0]; if (piv != c) swsign = -swsign; }
    __syncthreads();
    const int p = piv;
    if (p != c) {
      for (int cc = tid; cc < 512; cc += 256) {
        const float t = A[c * 512 + cc];
        A[c * 512 + cc] = A[p * 512 + cc];
        A[p * 512 + cc] = t;
      }
    }
    __syncthreads();
    const float inv = 1.0f / A[c * 512 + c];
    for (int rr = c + 1 + tid; rr < 512; rr += 256) {
      const float fac = A[rr * 512 + c] * inv;
      A[rr * 512 + c] = fac;
      for (int cc = c + 1; cc < 512; ++cc) A[rr * 512 + cc] -= fac * A[c * 512 + cc];
    }
    __syncthreads();
  }
  // reduce log|diag| and sign
  float lg = 0.0f; int ns = 0;
  for (int d = tid; d < 512; d += 256) {
    const float v = A[d * 512 + d];
    lg += __logf(fabsf(v));
    if (v < 0.0f) ns ^= 1;
  }
  sv[tid] = lg; si[tid] = ns;
  __syncthreads();
  for (int off = 128; off > 0; off >>= 1) {
    if (tid < off) { sv[tid] += sv[tid + off]; si[tid] ^= si[tid + off]; }
    __syncthreads();
  }
  if (tid == 0) {
    const int sgn = swsign * (si[0] ? -1 : 1);
    sld[k * 2 + 0] = (float)sgn;
    sld[k * 2 + 1] = sv[0];
  }
}

__global__ void k_final(const float* __restrict__ sld, float* __restrict__ out) {
  float mx = -1e30f;
  for (int k = 0; k < 4; ++k) mx = fmaxf(mx, sld[2 * k + 1]);
  float s = 0.0f;
  for (int k = 0; k < 4; ++k) s += sld[2 * k] * __expf(sld[2 * k + 1] - mx);
  out[0] = __logf(fabsf(s)) + mx;
}

// ---------------------------------------------------------------------------
extern "C" void kernel_launch(void* const* d_in, const int* in_sizes, int n_in,
                              void* d_out, int out_size, void* d_ws, size_t ws_size,
                              hipStream_t stream) {
  (void)in_sizes; (void)n_in; (void)out_size; (void)ws_size;
  const float* x       = (const float*)d_in[0];
  const float* fc1_w0  = (const float*)d_in[1];
  const float* fc1_b0  = (const float*)d_in[2];
  const float* fc1_w   = (const float*)d_in[3];
  const float* fc1_b   = (const float*)d_in[4];
  const float* fc2_w0  = (const float*)d_in[5];
  const float* fc2_b0  = (const float*)d_in[6];
  const float* fc2_w   = (const float*)d_in[7];
  const float* fc2_b   = (const float*)d_in[8];
  const float* final_w = (const float*)d_in[9];
  const float* final_b = (const float*)d_in[10];
  const float* w_det   = (const float*)d_in[11];
  const float* b_det   = (const float*)d_in[12];
  const float* alpha   = (const float*)d_in[13];

  char* ws = (char*)d_ws;
  size_t off = 0;
  auto take = [&](size_t bytes) -> char* {
    char* p = ws + off;
    off = (off + bytes + 255) & ~(size_t)255;
    return p;
  };
  float*     g2raw = (float*)take((size_t)3 * 1024 * 32 * 4);
  float*     g2h   = (float*)take((size_t)9 * 1024 * 16 * 4);
  float*     h1    = (float*)take((size_t)1024 * 64 * 4);
  _Float16*  h1h   = (_Float16*)take((size_t)1024 * 64 * 2);
  _Float16*  fb    = (_Float16*)take((size_t)1024 * 320 * 2);
  float*     g1    = (float*)take((size_t)192 * 4);
  float*     zb    = (float*)take((size_t)512 * 3 * 4);
  float*     rbuf  = (float*)take((size_t)512 * 512 * 4);
  _Float16*  tmph  = (_Float16*)take((size_t)4 * 512 * 64 * 2);
  float*     rbias = (float*)take((size_t)2048 * 4);
  float*     Am    = (float*)take((size_t)4 * 512 * 512 * 4);
  float*     sld   = (float*)take((size_t)8 * 4);
  // padded/transposed f16 weights
  _Float16*  wt2   = (_Float16*)take((size_t)3 * 16 * 32 * 2);    // h2 layers
  _Float16*  wt10  = (_Float16*)take((size_t)64 * 128 * 2);       // fc1_w0^T pad
  _Float16*  wt1   = (_Float16*)take((size_t)3 * 64 * 320 * 2);   // fc1_w^T pad
  _Float16*  wdT   = (_Float16*)take((size_t)4 * 64 * 64 * 2);    // w_det^T

  // weight prep (pad + transpose to f16)
  k_wpad<<<2, 256, 0, stream>>>(fc2_w0, wt2, 31, 16, 32);
  k_wpad<<<2, 256, 0, stream>>>(fc2_w, wt2 + 512, 16, 16, 32);
  k_wpad<<<2, 256, 0, stream>>>(fc2_w + 256, wt2 + 1024, 16, 16, 32);
  k_wpad<<<32, 256, 0, stream>>>(fc1_w0, wt10, 105, 64, 128);
  for (int d = 0; d < 3; ++d)
    k_wpad<<<80, 256, 0, stream>>>(fc1_w + (size_t)d * 304 * 64,
                                   wt1 + (size_t)d * 64 * 320, 304, 64, 320);
  for (int k = 0; k < 4; ++k)
    k_wpad<<<16, 256, 0, stream>>>(w_det + (size_t)k * 4096, wdT + (size_t)k * 4096,
                                   64, 64, 64);

  // zero the atomic accumulators
  k_zero<<<(3 * 1024 * 32 + 255) / 256, 256, 0, stream>>>(g2raw, 3 * 1024 * 32);
  k_zero<<<(9 * 1024 * 16 + 255) / 256, 256, 0, stream>>>(g2h, 9 * 1024 * 16);

  // fused pair features + h2 MLP (3 WMMA layers, fully in-register)
  k_h2<<<dim3(64, 32), 128, 0, stream>>>(x, wt2, fc2_b0, fc2_b, g2raw, g2h);

  // h1 chain
  k_buildf0<<<(1024 * 128) / 256, 256, 0, stream>>>(g2raw, fb);
  k_h1gemm<<<8, 256, 0, stream>>>(fb, 128, wt10, fc1_b0, h1, h1h, 0);
  for (int d = 1; d <= 3; ++d) {
    k_colmeans<<<1, 256, 0, stream>>>(h1, g1);
    k_buildf<<<(1024 * 320) / 256, 256, 0, stream>>>(
        h1, g1, g2h + (size_t)(d - 1) * 3 * 1024 * 16, fb);
    k_h1gemm<<<8, 256, 0, stream>>>(fb, 320, wt1 + (size_t)(d - 1) * 64 * 320,
                                    fc1_b + (d - 1) * 64, h1, h1h, 1);
  }

  // envelope distances
  k_z<<<2, 256, 0, stream>>>(h1, final_w, final_b, x, zb);
  k_r<<<(512 * 512) / 256, 256, 0, stream>>>(x, zb, rbuf);

  // determinant matrices: hu @ w_det, then @ hd^T fused with envelope
  k_tmp<<<dim3(4, 4), 256, 0, stream>>>(h1h, wdT, tmph);
  k_rowbias<<<8, 256, 0, stream>>>(h1, b_det, rbias);
  k_amat<<<dim3(32, 4), 256, 0, stream>>>(tmph, h1h, rbias, rbuf, alpha, Am);

  // slogdet + logsumexp combine
  k_lu<<<4, 256, 0, stream>>>(Am, sld);
  k_final<<<1, 1, 0, stream>>>(sld, (float*)d_out);
}
